// STGCNBlock_49864570306622
// MI455X (gfx1250) — compile-verified, matrix operand order checked by reading, not run on hardware
//
#include <hip/hip_runtime.h>

#define Bsz 8
#define Nn 10000
#define Tt 12
#define Ee 160000
#define NT 120000          // Nn*Tt
#define BT 96              // Bsz*Tt
#define ROWS 960000        // BT*Nn
#define NC 640000          // Nn*64, row stride of one bt-snapshot
#define EPSV 1e-5f

typedef __attribute__((ext_vector_type(16))) _Float16 v16h;
typedef __attribute__((ext_vector_type(8)))  float    v8f;
typedef __attribute__((ext_vector_type(4)))  float    f4;
typedef __attribute__((ext_vector_type(4)))  unsigned int u4;

union FragU {
  u4 q[2];
  v16h h;
};

// CDNA5 16-bit fragment: lane holds K = [8*khi, 8*khi+8) and [16+8*khi, 24+8*khi)
// from a K-contiguous base -> two ds_load_b128 per fragment.
__device__ __forceinline__ v16h ld_frag(const _Float16* base, int khi) {
  FragU u;
  u.q[0] = *(const u4*)(base + 8 * khi);
  u.q[1] = *(const u4*)(base + 16 + 8 * khi);
  return u.h;
}

__device__ __forceinline__ v16h ld_frag_pred(const _Float16* base, int khi,
                                             bool valid) {
  u4 z = {0u, 0u, 0u, 0u};
  u4 a = *(const u4*)(base + 8 * khi);
  u4 b = *(const u4*)(base + 16 + 8 * khi);
  FragU u;
  u.q[0] = valid ? a : z;  // lowers to v_cndmask, no exec branching
  u.q[1] = valid ? b : z;
  return u.h;
}

// fragment from contiguous f32 memory (global), converted to f16
__device__ __forceinline__ v16h ld_frag_f32(const float* base, int khi) {
  v16h r;
#pragma unroll
  for (int i = 0; i < 8; ++i) r[i] = (_Float16)base[8 * khi + i];
#pragma unroll
  for (int i = 0; i < 8; ++i) r[8 + i] = (_Float16)base[16 + 8 * khi + i];
  return r;
}

__device__ __forceinline__ unsigned pack2(float a, float b) {
  union {
    _Float16 h[2];
    unsigned u;
  } p;
  p.h[0] = (_Float16)a;
  p.h[1] = (_Float16)b;
  return p.u;
}

__device__ __forceinline__ v8f wmma16(v16h a, v16h b, v8f c) {
  return __builtin_amdgcn_wmma_f32_16x16x32_f16(false, a, false, b, (short)0, c,
                                                false, false);
}

// Guaranteed hardware f32 atomic add (no CAS loop)
__device__ __forceinline__ void atomic_add_f32(float* p, float v) {
  asm volatile("global_atomic_add_f32 %0, %1, off" ::"v"(p), "v"(v) : "memory");
}

// ---------------- graph normalization prep ----------------
__global__ __launch_bounds__(256) void deg_kernel(const int* __restrict__ ei,
                                                  const float* __restrict__ ew,
                                                  float* __restrict__ deg) {
  int e = blockIdx.x * 256 + threadIdx.x;
  if (e < Ee) atomic_add_f32(deg + ei[e], ew[e]);
}

__global__ __launch_bounds__(256) void dinv_kernel(const float* __restrict__ deg,
                                                   float* __restrict__ dinv) {
  int i = blockIdx.x * 256 + threadIdx.x;
  if (i < Nn) {
    float d = deg[i];
    dinv[i] = d > 0.f ? rsqrtf(d) : 0.f;
  }
}

__global__ __launch_bounds__(256) void norm_kernel(const int* __restrict__ ei,
                                                   const float* __restrict__ ew,
                                                   const float* __restrict__ dinv,
                                                   float* __restrict__ nrm) {
  int e = blockIdx.x * 256 + threadIdx.x;
  if (e < Ee) nrm[e] = -ew[e] * dinv[ei[e]] * dinv[ei[Ee + e]];
}

// ---------------- stage 1: conv1x3 + GLU + BN  (and fused 1x1 residual) ----
// x:(8,32,N,T)  -> h0:(96,N,64) [h0[b*12+t, n, c]] ; res -> d_out (8,64,N,T)
#define W1S 104  // padded K-stride (halves): 208B rows, conflict-free b128
#define WRS 40
#define X1S 40
__global__ __launch_bounds__(256) void stage_temporal1(
    const float* __restrict__ x, const float* __restrict__ wg,
    const float* __restrict__ bg, const float* __restrict__ bng,
    const float* __restrict__ bnb, const float* __restrict__ bnm,
    const float* __restrict__ bnv, const float* __restrict__ wr,
    const float* __restrict__ br, float* __restrict__ h0,
    float* __restrict__ res_out) {
  __shared__ __align__(16) _Float16 Wl[128 * W1S];  // [cout][K], K=tap*32+cin
  __shared__ __align__(16) _Float16 Wr[64 * WRS];   // [cout][cin]
  __shared__ __align__(16) _Float16 Xl[18 * X1S];   // [pos][cin], halo 18 pos
  __shared__ float Qs[64][16];                      // sigmoid(q) exchange
  __shared__ float Cb[128], Csc[64], Csh[64], Rb[64];

  const int tid = threadIdx.x, lane = tid & 31, wave = tid >> 5;
  const int ln15 = lane & 15, khi = lane >> 4;

  for (int i = tid; i < 128 * 48; i += 256) {  // pairs of K
    int cout = i / 48, k0 = (i % 48) * 2;
    int tap = k0 >> 5, cin = k0 & 31;
    float w0 = wg[(cout * 32 + cin) * 3 + tap];
    float w1 = wg[(cout * 32 + cin + 1) * 3 + tap];
    *(unsigned*)&Wl[cout * W1S + k0] = pack2(w0, w1);
  }
  for (int i = tid; i < 64 * 16; i += 256) {
    int cout = i / 16, c0 = (i % 16) * 2;
    *(unsigned*)&Wr[cout * WRS + c0] =
        pack2(wr[cout * 32 + c0], wr[cout * 32 + c0 + 1]);
  }
  if (tid < 128) Cb[tid] = bg[tid];
  if (tid < 64) {
    float sc = bng[tid] * rsqrtf(bnv[tid] + EPSV);
    Csc[tid] = sc;
    Csh[tid] = bnb[tid] - bnm[tid] * sc;
    Rb[tid] = br[tid];
  }
  __syncthreads();

  // per-wave weight fragments (wave w owns couts [16w,16w+16))
  v16h af[3];
#pragma unroll
  for (int s = 0; s < 3; ++s)
    af[s] = ld_frag(&Wl[(wave * 16 + ln15) * W1S + s * 32], khi);
  v16h arf = ld_frag(&Wr[((wave & 3) * 16 + ln15) * WRS], khi);

  for (int cbk = blockIdx.x; cbk < 8 * 7500; cbk += gridDim.x) {
    __syncthreads();
    int b = cbk / 7500, p0 = (cbk % 7500) * 16;
    for (int i = tid; i < 18 * 16; i += 256) {  // 18 pos x 16 cin-pairs
      int hh = i / 16, c0 = (i % 16) * 2;
      long q = (long)p0 - 1 + hh;
      bool ok = (q >= 0 && q < NT);
      float v0 = ok ? x[((long)b * 32 + c0) * NT + q] : 0.f;
      float v1 = ok ? x[((long)b * 32 + c0 + 1) * NT + q] : 0.f;
      *(unsigned*)&Xl[hh * X1S + c0] = pack2(v0, v1);
    }
    __syncthreads();

    int pcol = p0 + ln15, tcol = pcol % 12;
    v8f acc{};
    v16h bf1;
#pragma unroll
    for (int s = 0; s < 3; ++s) {
      bool valid = (tcol + s - 1 >= 0) && (tcol + s - 1 < 12);
      v16h bf = ld_frag_pred(&Xl[(ln15 + s) * X1S], khi, valid);
      if (s == 1) bf1 = bf;
      acc = wmma16(af[s], bf, acc);
    }

    if (wave < 4) {  // fused 1x1 residual projection (center tap columns)
      v8f racc{};
      racc = wmma16(arf, bf1, racc);
#pragma unroll
      for (int r = 0; r < 8; ++r) {
        int c = wave * 16 + r + 8 * khi;
        res_out[((long)b * 64 + c) * NT + pcol] = racc[r] + Rb[c];
      }
    }
    if (wave >= 4) {  // gate channels 64..127
#pragma unroll
      for (int r = 0; r < 8; ++r) {
        int cq = (wave - 4) * 16 + r + 8 * khi;
        float qv = acc[r] + Cb[64 + cq];
        Qs[cq][ln15] = 1.f / (1.f + __expf(-qv));
      }
    }
    __syncthreads();
    if (wave < 4) {  // value channels 0..63 : GLU + BN, write (BT,N,C) layout
      int nrow = pcol / 12;
      long base = (((long)b * 12 + tcol) * Nn + nrow) * 64;
#pragma unroll
      for (int r = 0; r < 8; ++r) {
        int c = wave * 16 + r + 8 * khi;
        float val = (acc[r] + Cb[c]) * Qs[c][ln15];
        h0[base + c] = val * Csc[c] + Csh[c];
      }
    }
  }
}

// ---------------- Lhat sparse scatter:  Out[:,col,:] += norm * Z[:,row,:] ---
__global__ __launch_bounds__(256) void lhat_kernel(const float* __restrict__ Z,
                                                   const int* __restrict__ ei,
                                                   const float* __restrict__ nrm,
                                                   float* __restrict__ Out) {
  int e = blockIdx.x;
  float w = nrm[e];
  if (w == 0.f) return;
  int row = ei[e], col = ei[Ee + e];
  int lane = threadIdx.x & 31, wave = threadIdx.x >> 5;
  for (int bt = wave; bt < BT; bt += 8) {
    long sbase = (long)bt * NC + (long)row * 64 + lane * 2;
    long dbase = (long)bt * NC + (long)col * 64 + lane * 2;
    float2 v = *(const float2*)(Z + sbase);
    atomic_add_f32(Out + dbase, w * v.x);
    atomic_add_f32(Out + dbase + 1, w * v.y);
  }
}

// ---------------- Cheb combine GEMM (K=192) + bias + LayerNorm + ReLU ------
// out_row = [TA | TB | TC]_row @ [W0-W2 ; W1 ; 2*W2]  (folds Tx2 = 2*TC - TA)
#define WES 200
__global__ __launch_bounds__(256) void stage_cheb(
    const float* __restrict__ TA, const float* __restrict__ TB, const float* TC,
    const float* __restrict__ cw, const float* __restrict__ cbias,
    const float* __restrict__ lg, const float* __restrict__ lb, float* Hout) {
  __shared__ __align__(16) _Float16 WeffT[64 * WES];  // [d][k], K-contiguous
  const int tid = threadIdx.x, lane = tid & 31, wave = tid >> 5;
  const int ln15 = lane & 15, khi = lane >> 4;

  for (int i = tid; i < 64 * 96; i += 256) {  // 64 d x 96 K-pairs
    int d = i / 96, k0 = (i % 96) * 2;
    int seg = k0 >> 6, c0 = k0 & 63;
    float w0, w1;
    if (seg == 0) {
      w0 = cw[c0 * 64 + d] - cw[2 * 4096 + c0 * 64 + d];
      w1 = cw[(c0 + 1) * 64 + d] - cw[2 * 4096 + (c0 + 1) * 64 + d];
    } else if (seg == 1) {
      w0 = cw[4096 + c0 * 64 + d];
      w1 = cw[4096 + (c0 + 1) * 64 + d];
    } else {
      w0 = 2.f * cw[2 * 4096 + c0 * 64 + d];
      w1 = 2.f * cw[2 * 4096 + (c0 + 1) * 64 + d];
    }
    *(unsigned*)&WeffT[d * WES + k0] = pack2(w0, w1);
  }
  __syncthreads();

  float lgv[4], lbv[4], cbv[4];
#pragma unroll
  for (int ct = 0; ct < 4; ++ct) {
    lgv[ct] = lg[ct * 16 + ln15];
    lbv[ct] = lb[ct * 16 + ln15];
    cbv[ct] = cbias[ct * 16 + ln15];
  }

  for (long blk = (long)blockIdx.x * 8 + wave; blk < 60000;
       blk += (long)gridDim.x * 8) {
    long row = blk * 16 + ln15;  // A-fragment lane owns data row (row mod 16)
    const float* rowp[3] = {TA + row * 64, TB + row * 64, TC + row * 64};
    v16h af[6];
#pragma unroll
    for (int s = 0; s < 6; ++s)
      af[s] = ld_frag_f32(rowp[s >> 1] + ((s & 1) << 5), khi);

    v8f acc[4];
#pragma unroll
    for (int ct = 0; ct < 4; ++ct) {
      v8f a{};
#pragma unroll
      for (int s = 0; s < 6; ++s) {
        v16h bf = ld_frag(&WeffT[(ct * 16 + ln15) * WES + s * 32], khi);
        a = wmma16(af[s], bf, a);
      }
#pragma unroll
      for (int r = 0; r < 8; ++r) a[r] += cbv[ct];
      acc[ct] = a;
    }
    // LayerNorm across 64 channels (16-lane groups hold the 16 cols of a tile)
    long rbase = blk * 16;
#pragma unroll
    for (int r = 0; r < 8; ++r) {
      float sv = 0.f, sq = 0.f;
#pragma unroll
      for (int ct = 0; ct < 4; ++ct) {
        float t = acc[ct][r];
        sv += t;
        sq += t * t;
      }
      for (int msk = 1; msk < 16; msk <<= 1) {
        sv += __shfl_xor(sv, msk, 32);
        sq += __shfl_xor(sq, msk, 32);
      }
      float mu = sv * (1.f / 64.f);
      float var = sq * (1.f / 64.f) - mu * mu;
      float rs = rsqrtf(var + EPSV);
      long orow = rbase + r + 8 * khi;
#pragma unroll
      for (int ct = 0; ct < 4; ++ct) {
        float val = (acc[ct][r] - mu) * rs * lgv[ct] + lbv[ct];
        Hout[orow * 64 + ct * 16 + ln15] = fmaxf(val, 0.f);
      }
    }
  }
}

// ---------------- stage 2: conv1x3 + GLU + BN + residual add ---------------
// hr:(96,N,64) -> d_out += conv  (d_out already holds residual)
#define W2S 200
#define X2S 72
__global__ __launch_bounds__(256) void stage_temporal2(
    const float* __restrict__ hr, const float* __restrict__ wg,
    const float* __restrict__ bg, const float* __restrict__ bng,
    const float* __restrict__ bnb, const float* __restrict__ bnm,
    const float* __restrict__ bnv, float* __restrict__ out) {
  __shared__ __align__(16) _Float16 Wl[128 * W2S];  // [cout][K], K=tap*64+cin
  __shared__ __align__(16) _Float16 Xl[18 * X2S];   // [pos][cin]
  __shared__ float Qs[64][16];
  __shared__ float Cb[128], Csc[64], Csh[64];

  const int tid = threadIdx.x, lane = tid & 31, wave = tid >> 5;
  const int ln15 = lane & 15, khi = lane >> 4;

  for (int i = tid; i < 128 * 96; i += 256) {  // pairs of K
    int cout = i / 96, k0 = (i % 96) * 2;
    int tap = k0 / 64, cin = k0 & 63;
    float w0 = wg[(cout * 64 + cin) * 3 + tap];
    float w1 = wg[(cout * 64 + cin + 1) * 3 + tap];
    *(unsigned*)&Wl[cout * W2S + k0] = pack2(w0, w1);
  }
  if (tid < 128) Cb[tid] = bg[tid];
  if (tid < 64) {
    float sc = bng[tid] * rsqrtf(bnv[tid] + EPSV);
    Csc[tid] = sc;
    Csh[tid] = bnb[tid] - bnm[tid] * sc;
  }
  __syncthreads();

  v16h af[6];
#pragma unroll
  for (int s = 0; s < 6; ++s)
    af[s] = ld_frag(&Wl[(wave * 16 + ln15) * W2S + s * 32], khi);

  for (int cbk = blockIdx.x; cbk < 8 * 7500; cbk += gridDim.x) {
    __syncthreads();
    int b = cbk / 7500, p0 = (cbk % 7500) * 16;
    for (int i = tid; i < 18 * 16; i += 256) {
      int hh = i / 16, off = (i & 15) * 4;
      long q = (long)p0 - 1 + hh;
      f4 v{};
      if (q >= 0 && q < NT) {
        int nq = (int)(q / 12), tq = (int)(q % 12);
        v = *(const f4*)(hr + (((long)b * 12 + tq) * Nn + nq) * 64 + off);
      }
      *(unsigned*)&Xl[hh * X2S + off] = pack2(v[0], v[1]);
      *(unsigned*)&Xl[hh * X2S + off + 2] = pack2(v[2], v[3]);
    }
    __syncthreads();

    int pcol = p0 + ln15, tcol = pcol % 12;
    v8f acc{};
#pragma unroll
    for (int s = 0; s < 6; ++s) {
      int tap = s >> 1, chalf = (s & 1) << 5;
      bool valid = (tcol + tap - 1 >= 0) && (tcol + tap - 1 < 12);
      v16h bf = ld_frag_pred(&Xl[(ln15 + tap) * X2S + chalf], khi, valid);
      acc = wmma16(af[s], bf, acc);
    }
    if (wave >= 4) {
#pragma unroll
      for (int r = 0; r < 8; ++r) {
        int cq = (wave - 4) * 16 + r + 8 * khi;
        float qv = acc[r] + Cb[64 + cq];
        Qs[cq][ln15] = 1.f / (1.f + __expf(-qv));
      }
    }
    __syncthreads();
    if (wave < 4) {
#pragma unroll
      for (int r = 0; r < 8; ++r) {
        int c = wave * 16 + r + 8 * khi;
        float val = (acc[r] + Cb[c]) * Qs[c][ln15];
        val = val * Csc[c] + Csh[c];
        long oidx = ((long)b * 64 + c) * NT + pcol;
        out[oidx] = val + out[oidx];  // add residual stored by stage 1
      }
    }
  }
}

extern "C" void kernel_launch(void* const* d_in, const int* in_sizes, int n_in,
                              void* d_out, int out_size, void* d_ws,
                              size_t ws_size, hipStream_t stream) {
  const float* x = (const float*)d_in[0];
  const int* ei = (const int*)d_in[1];
  const float* ew = (const float*)d_in[2];
  const float* wg1 = (const float*)d_in[3];
  const float* bg1 = (const float*)d_in[4];
  const float* bn1g = (const float*)d_in[5];
  const float* bn1b = (const float*)d_in[6];
  const float* bn1m = (const float*)d_in[7];
  const float* bn1v = (const float*)d_in[8];
  const float* cw = (const float*)d_in[9];
  const float* cb = (const float*)d_in[10];
  const float* lng = (const float*)d_in[11];
  const float* lnb = (const float*)d_in[12];
  const float* wg2 = (const float*)d_in[13];
  const float* bg2 = (const float*)d_in[14];
  const float* bn2g = (const float*)d_in[15];
  const float* bn2b = (const float*)d_in[16];
  const float* bn2m = (const float*)d_in[17];
  const float* bn2v = (const float*)d_in[18];
  const float* wres = (const float*)d_in[19];
  const float* bres = (const float*)d_in[20];

  float* out = (float*)d_out;
  const size_t SZ = (size_t)ROWS * 64;
  float* bufA = (float*)d_ws;  // Tx0 = h0
  float* bufB = bufA + SZ;     // Tx1
  float* bufC = bufB + SZ;     // Lhat(Tx1), then relu(LN(...)) in-place
  float* deg = bufC + SZ;
  float* dinv = deg + Nn;
  float* nrm = dinv + Nn;

  // graph normalization
  hipMemsetAsync(deg, 0, Nn * sizeof(float), stream);
  deg_kernel<<<(Ee + 255) / 256, 256, 0, stream>>>(ei, ew, deg);
  dinv_kernel<<<(Nn + 255) / 256, 256, 0, stream>>>(deg, dinv);
  norm_kernel<<<(Ee + 255) / 256, 256, 0, stream>>>(ei, ew, dinv, nrm);

  // temporal block 1 (+ residual into d_out)
  stage_temporal1<<<2048, 256, 0, stream>>>(x, wg1, bg1, bn1g, bn1b, bn1m,
                                            bn1v, wres, bres, bufA, out);

  // Chebyshev recursion via atomic scatter
  hipMemsetAsync(bufB, 0, SZ * sizeof(float), stream);
  lhat_kernel<<<Ee, 256, 0, stream>>>(bufA, ei, nrm, bufB);
  hipMemsetAsync(bufC, 0, SZ * sizeof(float), stream);
  lhat_kernel<<<Ee, 256, 0, stream>>>(bufB, ei, nrm, bufC);

  // combine + LayerNorm + ReLU (writes h back into bufC)
  stage_cheb<<<2048, 256, 0, stream>>>(bufA, bufB, bufC, cw, cb, lng, lnb,
                                       bufC);

  // temporal block 2 + residual add
  stage_temporal2<<<2048, 256, 0, stream>>>(bufC, wg2, bg2, bn2g, bn2b, bn2m,
                                            bn2v, out);
}
